// EResNetM_30683246362997
// MI455X (gfx1250) — matrix operand order repaired and checked by reading, not compiled, not run on hardware
//
#include <hip/hip_runtime.h>
#include <math.h>

// ---------------------------------------------------------------- constants
#define BATCH   4096
#define IN      256
#define SAMP    231     // sampled feature count
#define OUTN    32
#define NCOMP   64
#define BT      64      // batch rows per workgroup tile (4 m-tiles)
#define MT      (BT/16)
#define KPA     256     // padded K for gathered input (231 -> 256)
#define HMAX    512     // max supported hidden width (exp(64) tail clamp)
#define LN_EPS  1e-5f

typedef _Float16 half_t;
typedef __attribute__((ext_vector_type(16))) _Float16 v16h;
typedef __attribute__((ext_vector_type(8)))  float    v8f;
typedef __attribute__((ext_vector_type(4)))  float    v4f;

union V16H { v16h v; half_t h[16]; };
union H4U  { unsigned long long u; half_t h[4]; };

static __host__ __device__ inline int ceil16i(int x) { return (x + 15) & ~15; }
static __host__ __device__ inline int ceil32i(int x) { return (x + 31) & ~31; }

// dynamic LDS arena sizes (bytes)
#define COMP_SMEM (BT*KPA*2 + 2*BT*HMAX*2 + BT*OUTN*4 + 1024)   // 173,056 B
#define MGR_SMEM  (BT*IN*2 + 3*BT*64*2 + BT*64*4)               //  73,728 B

struct CompMeta {
    const half_t *Wp1, *Wp2, *WpfT, *WpfB;              // packed f16 fragments
    const float  *b1, *g1, *bt1, *b2, *g2, *bt2;
    const float  *bf, *gf, *btf;
    int h1, h2;
};
struct MetaChunk { CompMeta m[8]; };

struct MgrParams {
    const half_t *Wp1, *Wp2, *Wp3, *WpfA, *WpfB, *WpfC; // packed f16 fragments
    const float  *b1, *g1, *bt1, *b2, *g2, *bt2, *b3, *g3, *bt3;
    const float  *bf, *gf, *btf;
};

// ---------------------------------------------------------------- fragments
// A (16x32 f16) from LDS, documented CDNA5 layout:
//   lanes 0-15 : M=lane,    K in {0..7, 16..23}
//   lanes16-31 : M=lane-16, K in {8..15, 24..31}
__device__ inline v16h load_a_frag(const half_t* smA, int Kp, int lane, int ks) {
    int m  = lane & 15;
    int kh = (lane >> 4) * 8;
    const half_t* base = smA + m * Kp + ks * 32 + kh;
    V16H r;
    *(v4f*)&r.h[0] = *(const v4f*)(base);        // K = kh .. kh+7
    *(v4f*)&r.h[8] = *(const v4f*)(base + 16);   // K = kh+16 .. kh+23
    return r.v;
}

// One 16x16 output tile from pre-packed B fragments.
// Packed layout: fragment (nt,ks), lane l -> 16 contiguous halves at
//   Wp + ((nt*ksteps + ks)*32 + l) * 16
__device__ inline v8f gemm_tile_p(const half_t* smA, int Kp,
                                  const half_t* __restrict__ Wp, int nt, int lane) {
    v8f acc = {};
    int ksteps = Kp >> 5;
    const half_t* wp = Wp + ((size_t)nt * ksteps * 32 + lane) * 16;
    for (int ks = 0; ks < ksteps; ++ks) {
        v16h a = load_a_frag(smA, Kp, lane, ks);
        V16H b;
        *(v4f*)&b.h[0] = *(const v4f*)(wp);
        *(v4f*)&b.h[8] = *(const v4f*)(wp + 8);
        wp += 32 * 16;
        acc = __builtin_amdgcn_wmma_f32_16x16x32_f16(
                  false, a, false, b.v, (short)0, acc, false, false);
    }
    return acc;
}

// ---------------------------------------------------------------- layer driver
// smOut(BT x KpOut f16) = LayerNorm(relu(smA(BT x Kp) @ W + b)) ; padding zeroed.
// Wave mapping: wave -> (mt = wave>>1, n-group = wave&1), n-tiles strided by 2.
__device__ void hidden_layer(const half_t* smA, int Kp,
                             const half_t* __restrict__ Wp,
                             const float* __restrict__ b, const float* __restrict__ g,
                             const float* __restrict__ bt,
                             int N, half_t* smOut, int KpOut) {
    int tid = threadIdx.x, wave = tid >> 5, lane = tid & 31;
    int mt = wave >> 1, ng = wave & 1;

    for (int i = tid; i < (BT * KpOut) / 8; i += 256) {
        v4f z = {0.0f, 0.0f, 0.0f, 0.0f};
        *(v4f*)(smOut + i * 8) = z;
    }
    __syncthreads();

    int ntiles = (N + 15) >> 4;
    for (int nt = ng; nt < ntiles; nt += 2) {
        v8f acc = gemm_tile_p(smA + mt * 16 * Kp, Kp, Wp, nt, lane);
        int col   = nt * 16 + (lane & 15);
        int rbase = mt * 16 + (lane >> 4) * 8;   // C layout: lanes16-31 -> rows +8
        if (col < N) {
            float bias = b[col];
#pragma unroll
            for (int v = 0; v < 8; ++v) {
                float x = acc[v] + bias;
                x = fmaxf(x, 0.0f);
                smOut[(rbase + v) * KpOut + col] = (half_t)x;
            }
        }
    }
    __syncthreads();

    // LayerNorm: 8 rows per wave; stats over padded width (padding is exact zero)
    for (int r = wave * 8; r < wave * 8 + 8; ++r) {
        const half_t* rowp = smOut + r * KpOut;
        float s = 0.0f, s2 = 0.0f;
        for (int c0 = lane * 8; c0 < KpOut; c0 += 256) {
            V16H buf;
            *(v4f*)&buf.h[0] = *(const v4f*)(rowp + c0);
#pragma unroll
            for (int j = 0; j < 8; ++j) { float x = (float)buf.h[j]; s += x; s2 += x * x; }
        }
#pragma unroll
        for (int off = 16; off >= 1; off >>= 1) {
            s  += __shfl_xor(s,  off, 32);
            s2 += __shfl_xor(s2, off, 32);
        }
        float mean = s / (float)N;
        float var  = s2 / (float)N - mean * mean;
        float inv  = rsqrtf(var + LN_EPS);
        half_t* roww = smOut + r * KpOut;
        for (int c0 = lane * 8; c0 < N; c0 += 256) {
            if (c0 + 8 <= N) {
                V16H buf;
                *(v4f*)&buf.h[0] = *(const v4f*)(rowp + c0);
                v4f g0 = *(const v4f*)(g + c0),  g1 = *(const v4f*)(g + c0 + 4);
                v4f t0 = *(const v4f*)(bt + c0), t1 = *(const v4f*)(bt + c0 + 4);
#pragma unroll
                for (int j = 0; j < 8; ++j) {
                    float gg = (j < 4) ? g0[j] : g1[j - 4];
                    float tt = (j < 4) ? t0[j] : t1[j - 4];
                    float x  = (float)buf.h[j];
                    buf.h[j] = (half_t)((x - mean) * inv * gg + tt);
                }
                *(v4f*)(roww + c0) = *(v4f*)&buf.h[0];
            } else {
#pragma unroll
                for (int j = 0; j < 8; ++j) {
                    int c = c0 + j;
                    if (c < N) {
                        float x = (float)rowp[c];
                        roww[c] = (half_t)((x - mean) * inv * g[c] + bt[c]);
                    }
                }
            }
        }
    }
    __syncthreads();
}

// Accumulate smY(BT x Nout f32) += smA(BT x Kp) @ Wp. No barrier inside;
// same thread->element mapping on every call so repeated calls don't race.
__device__ void final_accum(const half_t* smA, int Kp,
                            const half_t* __restrict__ Wp, int Nout, float* smY) {
    int tid = threadIdx.x, wave = tid >> 5, lane = tid & 31;
    int mt = wave >> 1, ng = wave & 1;
    int ntiles = (Nout + 15) >> 4;
    for (int nt = ng; nt < ntiles; nt += 2) {
        v8f acc = gemm_tile_p(smA + mt * 16 * Kp, Kp, Wp, nt, lane);
        int col   = nt * 16 + (lane & 15);
        int rbase = mt * 16 + (lane >> 4) * 8;
#pragma unroll
        for (int v = 0; v < 8; ++v)
            smY[(rbase + v) * Nout + col] += acc[v];
    }
}

// ---------------------------------------------------------------- kernels
// Repack f32 row-major W(Kreal x N) into f16 WMMA-B fragment order, zero padded
// to (Kp x Np). One thread per fragment-lane (writes 16 contiguous halves).
__global__ void repack_kernel(const float* __restrict__ W, int Kreal, int N,
                              half_t* __restrict__ dst, int Kp, int Np) {
    int idx    = blockIdx.x * 256 + threadIdx.x;
    int ksteps = Kp >> 5;
    int total  = ksteps * (Np >> 4) * 32;
    if (idx >= total) return;
    int lane = idx & 31;
    int fs   = idx >> 5;            // nt*ksteps + ks
    int ks   = fs % ksteps;
    int nt   = fs / ksteps;
    int n    = nt * 16 + (lane & 15);
    int kb   = ks * 32 + (lane >> 4) * 8;
    V16H r;
#pragma unroll
    for (int j = 0; j < 16; ++j) {
        int k = kb + j + (j >= 8 ? 8 : 0);
        float w = (n < N && k < Kreal) ? W[k * N + n] : 0.0f;
        r.h[j] = (half_t)w;
    }
    half_t* out = dst + (size_t)idx * 16;
    *(v4f*)(out)      = *(v4f*)&r.h[0];
    *(v4f*)(out + 8)  = *(v4f*)&r.h[8];
}

__global__ void upload_meta_kernel(MetaChunk ch, CompMeta* dst, int base) {
    int i = threadIdx.x;
    if (i < 8) dst[base + i] = ch.m[i];
}

__global__ void zero_kernel(float* p, int n) {
    int i = blockIdx.x * 256 + threadIdx.x;
    if (i < n) p[i] = 0.0f;
}

__global__ void final_kernel(const float* __restrict__ acc, float* __restrict__ out, int n) {
    int i = blockIdx.x * 256 + threadIdx.x;
    if (i < n) out[i] = 1.0f / (1.0f + expf(-acc[i]));
}

__global__ __launch_bounds__(256)
void manager_kernel(const float* __restrict__ X, MgrParams p, float* __restrict__ gate) {
    extern __shared__ char smem[];
    half_t* smA  = (half_t*)smem;                               // BT*IN*2
    half_t* smH1 = (half_t*)(smem + BT*IN*2);                   // BT*64*2
    half_t* smH2 = (half_t*)(smem + BT*IN*2 + BT*64*2);
    half_t* smH3 = (half_t*)(smem + BT*IN*2 + 2*BT*64*2);
    float*  smY  = (float*)(smem + BT*IN*2 + 3*BT*64*2);        // BT*64*4

    int tid = threadIdx.x;
    int row0 = blockIdx.x * BT;

    for (int i = tid; i < (BT * IN) / 4; i += 256) {
        int e = i * 4;
        int r = e >> 8, col = e & 255;
        v4f x = *(const v4f*)(X + (row0 + r) * IN + col);
        H4U h;
#pragma unroll
        for (int j = 0; j < 4; ++j) h.h[j] = (half_t)x[j];
        *(unsigned long long*)(smA + e) = h.u;
    }
    __syncthreads();

    hidden_layer(smA,  IN, p.Wp1, p.b1, p.g1, p.bt1, 64, smH1, 64);
    hidden_layer(smH1, 64, p.Wp2, p.b2, p.g2, p.bt2, 64, smH2, 64);
    hidden_layer(smH2, 64, p.Wp3, p.b3, p.g3, p.bt3, 64, smH3, 64);

    for (int i = tid; i < (BT * 64) / 4; i += 256) {
        v4f z = {0.0f, 0.0f, 0.0f, 0.0f};
        *(v4f*)(smY + i * 4) = z;
    }
    __syncthreads();
    final_accum(smH1, 64, p.WpfA, 64, smY);
    final_accum(smH2, 64, p.WpfB, 64, smY);
    final_accum(smH3, 64, p.WpfC, 64, smY);
    __syncthreads();

    // LN over 64 cols + sigmoid -> gate. 16 threads per row, 4 cols each, 4 row-passes.
    int c16 = tid & 15;
    for (int r = tid >> 4; r < BT; r += 16) {
        float x[4], s = 0.0f, s2 = 0.0f;
#pragma unroll
        for (int j = 0; j < 4; ++j) {
            int col = c16 + j * 16;
            x[j] = smY[r * 64 + col] + p.bf[col];
            s += x[j]; s2 += x[j] * x[j];
        }
#pragma unroll
        for (int off = 8; off >= 1; off >>= 1) {
            s  += __shfl_xor(s,  off, 16);
            s2 += __shfl_xor(s2, off, 16);
        }
        float mean = s * (1.0f / 64.0f);
        float var  = s2 * (1.0f / 64.0f) - mean * mean;
        float inv  = rsqrtf(var + LN_EPS);
#pragma unroll
        for (int j = 0; j < 4; ++j) {
            int col = c16 + j * 16;
            float y = (x[j] - mean) * inv * p.gf[col] + p.btf[col];
            gate[(row0 + r) * NCOMP + col] = 1.0f / (1.0f + expf(-y));
        }
    }
}

__global__ __launch_bounds__(256)
void comp_kernel(const float* __restrict__ X, const int* __restrict__ feats,
                 const float* __restrict__ gate, float* __restrict__ acc,
                 const CompMeta* __restrict__ metas) {
    extern __shared__ char smem[];
    half_t* smA  = (half_t*)smem;                                   // BT*KPA*2
    half_t* smH1 = (half_t*)(smem + BT*KPA*2);                      // BT*HMAX*2
    half_t* smH2 = (half_t*)(smem + BT*KPA*2 + BT*HMAX*2);          // BT*HMAX*2
    float*  smY  = (float*)(smem + BT*KPA*2 + 2*BT*HMAX*2);         // BT*OUTN*4
    int* smFeat  = (int*)(smem + BT*KPA*2 + 2*BT*HMAX*2 + BT*OUTN*4);

    int c    = blockIdx.x;
    int row0 = blockIdx.y * BT;
    int tid  = threadIdx.x;

    CompMeta m = metas[c];
    int h1 = m.h1 < HMAX ? m.h1 : HMAX;
    int h2 = m.h2 < HMAX ? m.h2 : HMAX;
    int Kp1 = ceil32i(h1);
    int Kp2 = ceil32i(h2);

    for (int i = tid; i < SAMP; i += 256) smFeat[i] = feats[c * SAMP + i];
    __syncthreads();

    // gather X[:, feats[c]] -> f16 LDS tile, zero-padded K 231..255
    for (int i = tid; i < BT * KPA; i += 256) {
        int r = i >> 8, col = i & 255;
        float x = (col < SAMP) ? X[(row0 + r) * IN + smFeat[col]] : 0.0f;
        smA[i] = (half_t)x;
    }
    __syncthreads();

    hidden_layer(smA,  KPA, m.Wp1, m.b1, m.g1, m.bt1, h1, smH1, Kp1);
    hidden_layer(smH1, Kp1, m.Wp2, m.b2, m.g2, m.bt2, h2, smH2, Kp2);

    for (int i = tid; i < (BT * OUTN) / 4; i += 256) {
        v4f z = {0.0f, 0.0f, 0.0f, 0.0f};
        *(v4f*)(smY + i * 4) = z;
    }
    __syncthreads();
    // concat(H1,H2) @ Wf split along K into two packed matrices
    final_accum(smH1, Kp1, m.WpfT, OUTN, smY);
    final_accum(smH2, Kp2, m.WpfB, OUTN, smY);
    __syncthreads();

    // LN over 32 cols + gate scaling + global f32 atomic accumulate
    int c16 = tid & 15;
    for (int r = tid >> 4; r < BT; r += 16) {
        float x0 = smY[r * OUTN + c16]      + m.bf[c16];
        float x1 = smY[r * OUTN + c16 + 16] + m.bf[c16 + 16];
        float s = x0 + x1, s2 = x0 * x0 + x1 * x1;
#pragma unroll
        for (int off = 8; off >= 1; off >>= 1) {
            s  += __shfl_xor(s,  off, 16);
            s2 += __shfl_xor(s2, off, 16);
        }
        float mean = s * (1.0f / OUTN);
        float var  = s2 * (1.0f / OUTN) - mean * mean;
        float inv  = rsqrtf(var + LN_EPS);
        float y0 = (x0 - mean) * inv * m.gf[c16]      + m.btf[c16];
        float y1 = (x1 - mean) * inv * m.gf[c16 + 16] + m.btf[c16 + 16];
        float gv = gate[(row0 + r) * NCOMP + c] * (1.0f / (float)NCOMP);
        atomicAdd(&acc[(row0 + r) * OUTN + c16],      y0 * gv);
        atomicAdd(&acc[(row0 + r) * OUTN + c16 + 16], y1 * gv);
    }
}

// ---------------------------------------------------------------- launcher
static inline void launch_repack(const float* W, int Kreal, int N,
                                 half_t* dst, int Kp, int Np, hipStream_t stream) {
    int total = (Kp >> 5) * (Np >> 4) * 32;
    repack_kernel<<<(total + 255) / 256, 256, 0, stream>>>(W, Kreal, N, dst, Kp, Np);
}

extern "C" void kernel_launch(void* const* d_in, const int* in_sizes, int n_in,
                              void* d_out, int out_size, void* d_ws, size_t ws_size,
                              hipStream_t stream) {
    int idx = 0;
    const float* X = (const float*)d_in[idx++];

    struct RawComp { const float* p[12]; int h1, h2; } raw[NCOMP];
    for (int c = 0; c < NCOMP; ++c) {
        for (int j = 0; j < 12; ++j) { raw[c].p[j] = (const float*)d_in[idx];
            if (j == 1) raw[c].h1 = in_sizes[idx];
            if (j == 5) raw[c].h2 = in_sizes[idx];
            ++idx; }
    }
    const int* feats = (const int*)d_in[idx++];
    const float* mraw[16];
    for (int j = 0; j < 16; ++j) mraw[j] = (const float*)d_in[idx++];

    // d_ws layout: [acc][gate][CompMeta x64][packed f16 weights...]
    const size_t ACC_BYTES  = (size_t)BATCH * OUTN * sizeof(float);
    const size_t GATE_BYTES = (size_t)BATCH * NCOMP * sizeof(float);
    const size_t META_BYTES = ((sizeof(CompMeta) * NCOMP) + 255) & ~(size_t)255;
    float*    acc   = (float*)d_ws;
    float*    gate  = (float*)((char*)d_ws + ACC_BYTES);
    CompMeta* dmeta = (CompMeta*)((char*)d_ws + ACC_BYTES + GATE_BYTES);
    char*     packBase = (char*)d_ws + ACC_BYTES + GATE_BYTES + META_BYTES;

    size_t off = 0;
    CompMeta metas[NCOMP];
    for (int c = 0; c < NCOMP; ++c) {
        int h1 = raw[c].h1 < HMAX ? raw[c].h1 : HMAX;
        int h2 = raw[c].h2 < HMAX ? raw[c].h2 : HMAX;
        int Np1 = ceil16i(h1), Np2 = ceil16i(h2);
        int Kp2 = ceil32i(h1), KpfT = ceil32i(h1), KpfB = ceil32i(h2);
        CompMeta& m = metas[c];
        m.Wp1  = (half_t*)(packBase + off); off += (size_t)KPA  * Np1 * 2;
        m.Wp2  = (half_t*)(packBase + off); off += (size_t)Kp2  * Np2 * 2;
        m.WpfT = (half_t*)(packBase + off); off += (size_t)KpfT * OUTN * 2;
        m.WpfB = (half_t*)(packBase + off); off += (size_t)KpfB * OUTN * 2;
        m.b1 = raw[c].p[1];  m.g1 = raw[c].p[2];  m.bt1 = raw[c].p[3];
        m.b2 = raw[c].p[5];  m.g2 = raw[c].p[6];  m.bt2 = raw[c].p[7];
        m.bf = raw[c].p[9];  m.gf = raw[c].p[10]; m.btf = raw[c].p[11];
        m.h1 = h1; m.h2 = h2;
        // repack launches (weights are static across calls; recomputed deterministically)
        launch_repack(raw[c].p[0], SAMP, h1, (half_t*)m.Wp1, KPA,  Np1, stream);
        launch_repack(raw[c].p[4], h1,   h2, (half_t*)m.Wp2, Kp2,  Np2, stream);
        launch_repack(raw[c].p[8],                     h1, OUTN, (half_t*)m.WpfT, KpfT, OUTN, stream);
        launch_repack(raw[c].p[8] + (size_t)h1 * OUTN, h2, OUTN, (half_t*)m.WpfB, KpfB, OUTN, stream);
    }

    MgrParams mp;
    mp.Wp1  = (half_t*)(packBase + off); off += (size_t)IN * 64 * 2;
    mp.Wp2  = (half_t*)(packBase + off); off += (size_t)64 * 64 * 2;
    mp.Wp3  = (half_t*)(packBase + off); off += (size_t)64 * 64 * 2;
    mp.WpfA = (half_t*)(packBase + off); off += (size_t)64 * 64 * 2;
    mp.WpfB = (half_t*)(packBase + off); off += (size_t)64 * 64 * 2;
    mp.WpfC = (half_t*)(packBase + off); off += (size_t)64 * 64 * 2;
    mp.b1 = mraw[1];  mp.g1 = mraw[2];  mp.bt1 = mraw[3];
    mp.b2 = mraw[5];  mp.g2 = mraw[6];  mp.bt2 = mraw[7];
    mp.b3 = mraw[9];  mp.g3 = mraw[10]; mp.bt3 = mraw[11];
    mp.bf = mraw[13]; mp.gf = mraw[14]; mp.btf = mraw[15];
    launch_repack(mraw[0],  IN, 64, (half_t*)mp.Wp1,  IN, 64, stream);
    launch_repack(mraw[4],  64, 64, (half_t*)mp.Wp2,  64, 64, stream);
    launch_repack(mraw[8],  64, 64, (half_t*)mp.Wp3,  64, 64, stream);
    launch_repack(mraw[12],            64, 64, (half_t*)mp.WpfA, 64, 64, stream);
    launch_repack(mraw[12] + 64 * 64,  64, 64, (half_t*)mp.WpfB, 64, 64, stream);
    launch_repack(mraw[12] + 128 * 64, 64, 64, (half_t*)mp.WpfC, 64, 64, stream);

    // metadata to device via by-value kernel args (graph-capture safe, no memcpy)
    for (int ch = 0; ch < 8; ++ch) {
        MetaChunk chunk;
        for (int i = 0; i < 8; ++i) chunk.m[i] = metas[ch * 8 + i];
        upload_meta_kernel<<<1, 8, 0, stream>>>(chunk, dmeta, ch * 8);
    }

    zero_kernel<<<(BATCH * OUTN + 255) / 256, 256, 0, stream>>>(acc, BATCH * OUTN);
    manager_kernel<<<BATCH / BT, 256, MGR_SMEM, stream>>>(X, mp, gate);
    comp_kernel<<<dim3(NCOMP, BATCH / BT), 256, COMP_SMEM, stream>>>(X, feats, gate, acc, dmeta);
    final_kernel<<<(BATCH * OUTN + 255) / 256, 256, 0, stream>>>(acc, (float*)d_out, BATCH * OUTN);
}
